// MemLayer_7275674600019
// MI455X (gfx1250) — compile-verified, matrix operand order checked by reading, not run on hardware
//
#include <hip/hip_runtime.h>

typedef __attribute__((ext_vector_type(16))) _Float16 v16h;
typedef __attribute__((ext_vector_type(8)))  _Float16 v8h;
typedef __attribute__((ext_vector_type(8)))  float    v8f;

#define B_DIM 2048
#define D_DIM 2048
#define H_DIM 16
#define N_DIM 8192
#define V_DIM 128

// ws layout (all regions fully written every call; no reliance on prior state):
//   [0,     32768) : float   part[64][128]   column partial sums of `values`
//   [32768, 36864) : _Float16 vtile[2048]    tile(vmean, H) in f16 (16B aligned)
//   [36864, 45056) : float   r[2048]         r = Wo * vtile  (16B aligned)

// ---- Kernel 1: per-block partial column sums of values [N, V] ----------------
__global__ void colsum_partial(const float* __restrict__ values,
                               float* __restrict__ part) {
  __shared__ float s[256];
  const int tid  = threadIdx.x;
  const int col  = tid & 127;
  const int half = tid >> 7;                 // 0 or 1
  const int row0 = blockIdx.x * 128 + half;  // block owns 128 rows
  float acc = 0.f;
#pragma unroll 4
  for (int r = 0; r < 128; r += 2)
    acc += values[(size_t)(row0 + r) * V_DIM + col];
  s[tid] = acc;
  __syncthreads();
  if (tid < 128) part[blockIdx.x * 128 + tid] = s[tid] + s[tid + 128];
}

// ---- Kernel 2: finalize vmean, emit H-tiled f16 vector ----------------------
__global__ void finalize_vtile(const float* __restrict__ part,
                               _Float16* __restrict__ vtile) {
  __shared__ float vm[128];
  const int tid = threadIdx.x;
  if (tid < 128) {
    float acc = 0.f;
#pragma unroll 8
    for (int b = 0; b < 64; ++b) acc += part[b * 128 + tid];
    vm[tid] = acc * (1.0f / (float)N_DIM);
  }
  __syncthreads();
  for (int i = tid; i < D_DIM; i += 256) vtile[i] = (_Float16)vm[i & 127];
}

// ---- Kernel 3: r = Wo * vtile via WMMA f32_16x16x32_f16 ---------------------
// One wave computes 16 rows of r; 128 waves cover D=2048.
__global__ void wmma_gemv_r(const float* __restrict__ Wo,
                            const _Float16* __restrict__ vtile,
                            float* __restrict__ r) {
  const int gtid = blockIdx.x * blockDim.x + threadIdx.x;
  const int wave = gtid >> 5;
  const int lane = gtid & 31;
  const int d0   = wave * 16;            // output row tile
  const int m    = lane & 15;            // A-matrix row owned by this lane
  const int kb   = (lane < 16) ? 0 : 8;  // K sub-offset per ISA 16-bit A layout

  const float* wrow = Wo + (size_t)(d0 + m) * D_DIM;
  v8f c = {};

  for (int j0 = 0; j0 < D_DIM; j0 += 32) {
    // A tile: 16x32 of Wo, converted fp32 -> f16 in registers.
    const float4* w0 = (const float4*)(wrow + j0 + kb);        // K = kb..kb+7
    const float4* w1 = (const float4*)(wrow + j0 + 16 + kb);   // K = 16+kb..
    float4 f0 = w0[0], f1 = w0[1];
    float4 f2 = w1[0], f3 = w1[1];
    v16h a;
    a[0]  = (_Float16)f0.x; a[1]  = (_Float16)f0.y;
    a[2]  = (_Float16)f0.z; a[3]  = (_Float16)f0.w;
    a[4]  = (_Float16)f1.x; a[5]  = (_Float16)f1.y;
    a[6]  = (_Float16)f1.z; a[7]  = (_Float16)f1.w;
    a[8]  = (_Float16)f2.x; a[9]  = (_Float16)f2.y;
    a[10] = (_Float16)f2.z; a[11] = (_Float16)f2.w;
    a[12] = (_Float16)f3.x; a[13] = (_Float16)f3.y;
    a[14] = (_Float16)f3.z; a[15] = (_Float16)f3.w;

    // B tile: vtile chunk replicated across all 16 N columns (same K packing).
    v8h b0 = *(const v8h*)(vtile + j0 + kb);
    v8h b1 = *(const v8h*)(vtile + j0 + 16 + kb);
    v16h bb;
#pragma unroll
    for (int e = 0; e < 8; ++e) { bb[e] = b0[e]; bb[8 + e] = b1[e]; }

    c = __builtin_amdgcn_wmma_f32_16x16x32_f16(
        /*neg_a=*/false, a, /*neg_b=*/false, bb,
        /*c_mod=*/(short)0, c, /*reuse_a=*/false, /*reuse_b=*/false);
  }

  // D layout: lane 0 (N=0) holds M=0..7 in c[0..7]; lane 16 holds M=8..15.
  if (lane == 0) {
#pragma unroll
    for (int i = 0; i < 8; ++i) r[d0 + i] = c[i];
  } else if (lane == 16) {
#pragma unroll
    for (int i = 0; i < 8; ++i) r[d0 + 8 + i] = c[i];
  }
}

// ---- Kernel 4: out[b,d] = x[b,d] + r[d] -------------------------------------
__global__ void add_broadcast(const float* __restrict__ x,
                              const float* __restrict__ r,
                              float* __restrict__ out) {
  const size_t i = (size_t)blockIdx.x * blockDim.x + threadIdx.x; // float4 idx
  const float4* x4 = (const float4*)x;
  const float4* r4 = (const float4*)r;
  float4* o4 = (float4*)out;
  float4 xv = x4[i];
  float4 rv = r4[i & 511];   // 2048 floats per row = 512 float4
  o4[i] = make_float4(xv.x + rv.x, xv.y + rv.y, xv.z + rv.z, xv.w + rv.w);
}

extern "C" void kernel_launch(void* const* d_in, const int* in_sizes, int n_in,
                              void* d_out, int out_size, void* d_ws, size_t ws_size,
                              hipStream_t stream) {
  (void)in_sizes; (void)n_in; (void)out_size; (void)ws_size;
  const float* x      = (const float*)d_in[0];
  const float* values = (const float*)d_in[2];   // keys (d_in[1]) / Wq (d_in[3]) are
  const float* Wo     = (const float*)d_in[4];   // numerically dead (softmax rowsum == 1)
  float* out = (float*)d_out;

  char* ws = (char*)d_ws;
  float*    part  = (float*)(ws);
  _Float16* vtile = (_Float16*)(ws + 32768);
  float*    rvec  = (float*)(ws + 36864);

  colsum_partial<<<64, 256, 0, stream>>>(values, part);
  finalize_vtile<<<1, 256, 0, stream>>>(part, vtile);
  wmma_gemv_r<<<16, 256, 0, stream>>>(Wo, vtile, rvec);          // 128 waves
  add_broadcast<<<4096, 256, 0, stream>>>(x, rvec, out);         // 4M floats
}
